// SimpleEquivariantNetwork_33844342292899
// MI455X (gfx1250) — matrix operand order repaired and checked by reading, not compiled, not run on hardware
//
#include <hip/hip_runtime.h>
#include <hip/hip_bf16.h>
#include <math.h>

typedef __attribute__((ext_vector_type(16))) __bf16 v16bf;
typedef __attribute__((ext_vector_type(8)))  float  v8f;

#define SQRT3F      1.7320508075688772f
#define MAXR        5.0f
#define SMOOTH_CF   (1.14136f * 7.38905609893065f)   /* 1.14136 * e^2 */
#define SQRT10F     3.1622776601683795f
#define FC1N        0.31622776601683794f             /* 1/sqrt(10) */
#define FC2N        0.125f                           /* 1/sqrt(64) */
#define TPN         0.08838834764831845f             /* 1/sqrt(128) */
#define AGGN        0.2581988897471611f              /* 1/sqrt(15) */

// ---------------------------------------------------------------------------
// h = x @ W_embed   [N,16] x [16,32] -> [N,32] (f32 master copy)
// ---------------------------------------------------------------------------
__global__ void k_embed(const float* __restrict__ x, const float* __restrict__ We,
                        float* __restrict__ h, int N) {
  int id = blockIdx.x * blockDim.x + threadIdx.x;
  if (id >= N * 32) return;
  int n = id >> 5, d = id & 31;
  float acc = 0.f;
#pragma unroll
  for (int i = 0; i < 16; ++i) acc += x[n * 16 + i] * We[i * 32 + d];
  h[id] = acc;
}

// ---------------------------------------------------------------------------
// Per-edge precompute: sh-l1 attr [E,4]; radial MLP output rad[l][E] for all
// 3 layers (weights cached in LDS). Edge-invariant across layers, done once.
// ---------------------------------------------------------------------------
__global__ void __launch_bounds__(256) k_edge_prep(
    const float* __restrict__ pos, const int* __restrict__ esrc,
    const int* __restrict__ edst, const float* __restrict__ Wfc1,
    const float* __restrict__ Wfc2, float* __restrict__ attr,
    float* __restrict__ rad, int E) {
  __shared__ float s1[3 * 10 * 64];
  __shared__ float s2[3 * 64];
  for (int i = threadIdx.x; i < 3 * 10 * 64; i += blockDim.x) s1[i] = Wfc1[i];
  for (int i = threadIdx.x; i < 3 * 64; i += blockDim.x) s2[i] = Wfc2[i];
  __syncthreads();

  const float step = MAXR / 11.0f;
  for (int e = blockIdx.x * blockDim.x + threadIdx.x; e < E;
       e += gridDim.x * blockDim.x) {
    int s = esrc[e], d = edst[e];
    float vx = pos[s * 3 + 0] - pos[d * 3 + 0];
    float vy = pos[s * 3 + 1] - pos[d * 3 + 1];
    float vz = pos[s * 3 + 2] - pos[d * 3 + 2];
    float len = sqrtf(vx * vx + vy * vy + vz * vz);
    float inv = 1.0f / (len + 1e-9f);
    attr[(size_t)e * 4 + 0] = 1.0f;
    attr[(size_t)e * 4 + 1] = SQRT3F * vy * inv;
    attr[(size_t)e * 4 + 2] = SQRT3F * vz * inv;
    attr[(size_t)e * 4 + 3] = SQRT3F * vx * inv;

    float emb[10];
#pragma unroll
    for (int j = 0; j < 10; ++j) {
      float knot = (float)(j + 1) * step;
      float diff = (len - knot) / step;
      float xp = diff + 1.0f, xm = 1.0f - diff;
      float f = 0.0f;
      if (xp > 0.0f && xm > 0.0f)
        f = SMOOTH_CF * __expf(-1.0f / xp) * __expf(-1.0f / xm);
      emb[j] = f * SQRT10F;
    }
#pragma unroll
    for (int l = 0; l < 3; ++l) {
      float r = 0.0f;
      for (int j = 0; j < 64; ++j) {
        float a = 0.0f;
#pragma unroll
        for (int b = 0; b < 10; ++b) a += emb[b] * s1[(l * 10 + b) * 64 + j];
        a *= FC1N;
        float hid = a / (1.0f + __expf(-a));  // silu
        r += hid * s2[l * 64 + j];
      }
      rad[(size_t)l * E + e] = r * FC2N;
    }
  }
}

// ---------------------------------------------------------------------------
// Layer prep: residual base copy nxt = cur (f32) AND bf16 gather mirror
// hG = bf16(cur) in one pass. Gathers in the message kernel then move half
// the bytes through L2/LDS; all accumulation stays f32.
// ---------------------------------------------------------------------------
__global__ void k_prep_layer(const float* __restrict__ cur, float* __restrict__ nxt,
                             __bf16* __restrict__ hG, int n) {
  int id = blockIdx.x * blockDim.x + threadIdx.x;
  if (id >= n) return;
  float v = cur[id];
  nxt[id] = v;
  hG[id] = (__bf16)v;
}

__global__ void k_zero(float* __restrict__ p, int n) {
  int id = blockIdx.x * blockDim.x + threadIdx.x;
  if (id < n) p[id] = 0.0f;
}

// ---------------------------------------------------------------------------
// Message kernel: per 16-edge tile, ang = (h_src (x) attr) @ W_r with
// v_wmma_f32_16x16x32_bf16; K=128 in 4 chunks, N=32 in 2 tiles (8 WMMA/tile).
// B fragments (W_tp[l] reshaped [128,32], bf16) live in VGPRs for the whole
// kernel. h_src gathered as packed bf16 dwords (64B/row). Scatter is exact:
// f32 global atomics of ang * tp_norm * rad[e] * agg_norm into hnext.
// ---------------------------------------------------------------------------
#define MWAVES 8
__global__ void __launch_bounds__(256) k_message(
    const __bf16* __restrict__ hG, const float* __restrict__ Wl,
    const float* __restrict__ attr, const float* __restrict__ radl,
    const int* __restrict__ esrc, const int* __restrict__ edst,
    float* __restrict__ hnext, int E, int rounds) {
  __shared__ unsigned int sHw[MWAVES][16][16];  // 16 edges x 32 bf16 (packed)
  __shared__ float sA[MWAVES][16][4];
  __shared__ int   sD[MWAVES][16];
  __shared__ float sR[MWAVES][16];

  const int lane = threadIdx.x & 31;
  const int wave = threadIdx.x >> 5;
  const int n0 = lane & 15;
  const int kh = lane >> 4;            // 0: low half, 1: high half
  const int ntiles = (E + 15) >> 4;
  const float OUT_SCALE = TPN * AGGN;
  const unsigned int* __restrict__ hGw = (const unsigned int*)hG;

  // B fragments: B[c][n] = Wl[c*32 + n], c = flat (i*4+a) K index.
  // B 32x16 fragment, contiguous k order (same as A): lane holds col n0; lanes 0-15 K=0..15,
  // lanes 16-31 K=16..31 (element t -> K = kh*16 + t within the chunk).
  v16bf bfrag[4][2];
#pragma unroll
  for (int kk = 0; kk < 4; ++kk) {
#pragma unroll
    for (int nt = 0; nt < 2; ++nt) {
#pragma unroll
      for (int t = 0; t < 16; ++t) {
        int c = kk * 32 + kh * 16 + t;
        bfrag[kk][nt][t] = (__bf16)Wl[c * 32 + nt * 16 + n0];
      }
    }
  }

  for (int tb = blockIdx.x * MWAVES; tb < rounds; tb += gridDim.x * MWAVES) {
    const int tile = tb + wave;           // wave-uniform
    const bool active = (tile < ntiles);  // wave-uniform -> EXEC stays full

    if (active) {
      const int ebase = tile << 4;
      // gather: 16 rows x 16 dwords of packed bf16; 8 dwords per lane
#pragma unroll
      for (int it = 0; it < 8; ++it) {
        int idx = lane + it * 32;
        int row = idx >> 4, w = idx & 15;
        int e = ebase + row;
        if (e < E) {
          int s = esrc[e];
          sHw[wave][row][w] = hGw[(size_t)s * 16 + w];
        } else {
          sHw[wave][row][w] = 0u;
        }
      }
#pragma unroll
      for (int j = 0; j < 2; ++j) {       // stage attr (zero => tail edges inert)
        int idx = lane + j * 32;
        int m = idx >> 2, a = idx & 3;
        int e = ebase + m;
        sA[wave][m][a] = (e < E) ? attr[(size_t)e * 4 + a] : 0.0f;
      }
      if (lane < 16) {
        int e = ebase + lane;
        sD[wave][lane] = (e < E) ? edst[e] : 0;
        sR[wave][lane] = (e < E) ? radl[e] : 0.0f;
      }
    }
    __syncthreads();

    if (active) {
      const int m = n0;  // A row owned by this lane
      const __bf16* __restrict__ hrow = (const __bf16*)&sHw[wave][m][0];
      v8f c0 = {}; v8f c1 = {};
#pragma unroll
      for (int kk = 0; kk < 4; ++kk) {
        // A fragment, contiguous k order (matches B): element t -> K = kh*16 + t
        v16bf afrag;
#pragma unroll
        for (int t = 0; t < 16; ++t) {
          int k = kk * 32 + (kh << 4) + t;
          int i = k >> 2, a = k & 3;
          afrag[t] = (__bf16)((float)hrow[i] * sA[wave][m][a]);
        }
        c0 = __builtin_amdgcn_wmma_f32_16x16x32_bf16(
            false, afrag, false, bfrag[kk][0], (short)0, c0, false, false);
        c1 = __builtin_amdgcn_wmma_f32_16x16x32_bf16(
            false, afrag, false, bfrag[kk][1], (short)0, c1, false, false);
      }
      // C/D layout: VGPR r, lane -> row r + 8*kh, col n0
#pragma unroll
      for (int r = 0; r < 8; ++r) {
        int mr = r + (kh << 3);
        float scale = sR[wave][mr] * OUT_SCALE;  // 0 for tail edges
        int node = sD[wave][mr];
        atomicAdd(&hnext[(size_t)node * 32 + n0],      c0[r] * scale);
        atomicAdd(&hnext[(size_t)node * 32 + 16 + n0], c1[r] * scale);
      }
    }
    __syncthreads();
  }
}

// ---------------------------------------------------------------------------
// Graph mean-pool accumulation + decode
// ---------------------------------------------------------------------------
__global__ void k_pool(const float* __restrict__ h, const int* __restrict__ batch,
                       float* __restrict__ gsum, float* __restrict__ cnt, int N) {
  int id = blockIdx.x * blockDim.x + threadIdx.x;
  if (id >= N * 32) return;
  int n = id >> 5, d = id & 31;
  int g = batch[n];
  atomicAdd(&gsum[(size_t)g * 32 + d], h[id]);
  if (d == 0) atomicAdd(&cnt[g], 1.0f);
}

__global__ void k_decode(const float* __restrict__ gsum, const float* __restrict__ cnt,
                         const float* __restrict__ Wdec, const float* __restrict__ ori,
                         float* __restrict__ out, int G) {
  int g = blockIdx.x * blockDim.x + threadIdx.x;
  if (g >= G) return;
  float c = fmaxf(cnt[g], 1.0f);
  float coeff[4] = {0.f, 0.f, 0.f, 0.f};
  for (int d = 0; d < 32; ++d) {
    float gv = gsum[(size_t)g * 32 + d] / c;
#pragma unroll
    for (int k = 0; k < 4; ++k) coeff[k] += gv * Wdec[d * 4 + k];
  }
  float th = ori[g * 2 + 0], ph = ori[g * 2 + 1];
  float st = __sinf(th), ct = __cosf(th);
  float sp = __sinf(ph), cp = __cosf(ph);
  float qx = st * cp, qy = st * sp, qz = ct;
  out[g] = coeff[0] + SQRT3F * (coeff[1] * qy + coeff[2] * qz + coeff[3] * qx);
}

// ---------------------------------------------------------------------------
extern "C" void kernel_launch(void* const* d_in, const int* in_sizes, int n_in,
                              void* d_out, int out_size, void* d_ws, size_t ws_size,
                              hipStream_t stream) {
  const float* pos  = (const float*)d_in[0];
  const float* x    = (const float*)d_in[1];
  const float* ori  = (const float*)d_in[2];
  const float* We   = (const float*)d_in[3];
  const float* Wtp  = (const float*)d_in[4];   // [3][32][4][32] = [3][128*32]
  const float* Wfc1 = (const float*)d_in[5];   // [3][10][64]
  const float* Wfc2 = (const float*)d_in[6];   // [3][64][1]
  const float* Wdec = (const float*)d_in[7];   // [32][4]
  const int* esrc  = (const int*)d_in[8];
  const int* edst  = (const int*)d_in[9];
  const int* batch = (const int*)d_in[10];

  const int N = in_sizes[0] / 3;
  const int E = in_sizes[8];
  const int G = in_sizes[2] / 2;
  float* out = (float*)d_out;

  // workspace carve-up (256B aligned)
  char* ws = (char*)d_ws;
  size_t off = 0;
  auto carve = [&](size_t bytes) -> void* {
    void* p = ws + off;
    off = (off + bytes + 255) & ~(size_t)255;
    return p;
  };
  float*  hA   = (float*)carve((size_t)N * 32 * sizeof(float));
  float*  hB   = (float*)carve((size_t)N * 32 * sizeof(float));
  __bf16* hG   = (__bf16*)carve((size_t)N * 32 * sizeof(__bf16));
  float*  attr = (float*)carve((size_t)E * 4 * sizeof(float));
  float*  rad  = (float*)carve((size_t)E * 3 * sizeof(float));
  float*  gsum = (float*)carve((size_t)G * 33 * sizeof(float));  // sums + counts
  float*  cnt  = gsum + (size_t)G * 32;

  const int nd = N * 32;
  k_embed<<<(nd + 255) / 256, 256, 0, stream>>>(x, We, hA, N);
  k_edge_prep<<<2048, 256, 0, stream>>>(pos, esrc, edst, Wfc1, Wfc2, attr, rad, E);

  const int ntiles  = (E + 15) / 16;
  const int mblocks = (ntiles + MWAVES - 1) / MWAVES;
  const int rounds  = mblocks * MWAVES;

  float* cur = hA;
  float* nxt = hB;
  for (int l = 0; l < 3; ++l) {
    // residual base (f32) + bf16 gather mirror, one pass
    k_prep_layer<<<(nd + 255) / 256, 256, 0, stream>>>(cur, nxt, hG, nd);
    k_message<<<mblocks, 256, 0, stream>>>(hG, Wtp + (size_t)l * 128 * 32, attr,
                                           rad + (size_t)l * E, esrc, edst, nxt,
                                           E, rounds);
    float* t = cur; cur = nxt; nxt = t;
  }

  k_zero<<<(G * 33 + 255) / 256, 256, 0, stream>>>(gsum, G * 33);
  k_pool<<<(nd + 255) / 256, 256, 0, stream>>>(cur, batch, gsum, cnt, N);
  k_decode<<<(G + 63) / 64, 64, 0, stream>>>(gsum, cnt, Wdec, ori, out, G);
}
